// TransformerEncoderLayer_54451595379221
// MI455X (gfx1250) — matrix operand order repaired and checked by reading, not compiled, not run on hardware
//
#include <hip/hip_runtime.h>
#include <hip/hip_bf16.h>

// ---------------------------------------------------------------------------
// Types for CDNA5 WMMA (wave32, v_wmma_f32_16x16x32_bf16)
// ---------------------------------------------------------------------------
typedef __attribute__((ext_vector_type(16))) __bf16 v16bf;
typedef __attribute__((ext_vector_type(8)))  __bf16 v8bf;
typedef __attribute__((ext_vector_type(8)))  float  v8f;

#define DEV __device__ __forceinline__

DEV unsigned short f2bf(float f) {
    union { float f; unsigned u; } v; v.f = f;
    unsigned u = v.u;
    return (unsigned short)((u + 0x7FFFu + ((u >> 16) & 1u)) >> 16);
}

// A-fragment (16x32 bf16): lane m = l&15, kbase = (l>>4)*8.
// VGPR0-3 <-> K = kbase..kbase+7, VGPR4-7 <-> K = kbase+16..kbase+23.
DEV v16bf frag_a(const unsigned short* p) {
    v8bf lo = *(const v8bf*)(p);
    v8bf hi = *(const v8bf*)(p + 16);
    return __builtin_shufflevector(lo, hi, 0,1,2,3,4,5,6,7,8,9,10,11,12,13,14,15);
}

// B-fragment (32x16 bf16): lane n = l&15, kbase = (l>>4)*8.
// VGPR0-3 <-> K = kbase..kbase+7, VGPR4-7 <-> K = kbase+16..kbase+23.
DEV v16bf frag_b(const unsigned short* p) {
    v8bf lo = *(const v8bf*)(p);
    v8bf hi = *(const v8bf*)(p + 16);
    return __builtin_shufflevector(lo, hi, 0,1,2,3,4,5,6,7,8,9,10,11,12,13,14,15);
}

DEV v8f wmma_bf16(v16bf a, v16bf b, v8f c) {
    return __builtin_amdgcn_wmma_f32_16x16x32_bf16(
        false, a, false, b, (short)0, c, false, false);
}

// Async global->LDS copy of a 32-byte strip per lane (2x B128, ASYNCcnt +=2
// per call-site pair). INST_OFFSET applies to BOTH the LDS and global address
// (CDNA5 ISA), so offset:16 fetches the second 16B chunk.
DEV void async_copy_tile(unsigned ldsa, unsigned long long ga,
                         unsigned ldsb, unsigned long long gb) {
    asm volatile(
        "global_load_async_to_lds_b128 %0, %1, off\n\t"
        "global_load_async_to_lds_b128 %0, %1, off offset:16\n\t"
        "global_load_async_to_lds_b128 %2, %3, off\n\t"
        "global_load_async_to_lds_b128 %2, %3, off offset:16"
        :: "v"(ldsa), "v"(ga), "v"(ldsb), "v"(gb)
        : "memory");
}

// ---------------------------------------------------------------------------
// WMMA GEMM: C[M,N] = A[M,K] (bf16 row-major) x Bt[N,K] (bf16, pre-transposed)
// Block: 256 threads = 8 waves, 128x128 C tile; wave = 32x64 (2x4 accums,
// 8 WMMAs per K-stage). Tiles staged via GLOBAL_LOAD_ASYNC_TO_LDS_B128 with
// double buffering: issue stage j+1, s_wait_asynccnt 4 (stage j landed,
// in-order completion), barrier, compute stage j while j+1 streams in.
// M%128==0, N%128==0, K%32==0 (true for all GEMMs in this layer).
// ---------------------------------------------------------------------------
template<bool OUT_BF16, bool ACCUM>
__global__ void __launch_bounds__(256)
wmma_gemm(const unsigned short* __restrict__ A,
          const unsigned short* __restrict__ Bt,
          float* __restrict__ Cf,
          unsigned short* __restrict__ Cb,
          int M, int N, int K) {
    __shared__ alignas(16) unsigned short As[2][128][40];  // 80B rows: 16B-aligned cols
    __shared__ alignas(16) unsigned short Bs[2][128][40];

    const int tid  = threadIdx.x;
    const int lane = tid & 31;
    const int wave = tid >> 5;
    const int wm   = wave >> 1;     // 0..3  -> 32 rows
    const int wn   = wave & 1;      // 0..1  -> 64 cols
    const int m0   = blockIdx.x * 128;
    const int n0   = blockIdx.y * 128;
    const int lm   = lane & 15;
    const int hh   = lane >> 4;
    const int kbA  = hh * 8;        // A-fragment K offset
    const int kbB  = hh * 8;        // B-fragment K offset

    v8f acc[2][4];
#pragma unroll
    for (int i = 0; i < 2; ++i)
#pragma unroll
        for (int j = 0; j < 4; ++j)
#pragma unroll
            for (int r = 0; r < 8; ++r) {
                if (ACCUM) {
                    int row = m0 + wm * 32 + i * 16 + r + 8 * hh;
                    int col = n0 + wn * 64 + j * 16 + lm;
                    acc[i][j][r] = Cf[(size_t)row * N + col];
                } else {
                    acc[i][j][r] = 0.0f;
                }
            }

    // staging map: 256 threads cover 128 rows x 32 cols, 32B strip per thread
    const int srow = tid >> 1;           // 0..127
    const int scol = (tid & 1) * 16;     // 0,16

    // LDS byte offsets (low 32 bits of the generic shared pointer)
    unsigned lA[2] = { (unsigned)(size_t)&As[0][srow][scol],
                       (unsigned)(size_t)&As[1][srow][scol] };
    unsigned lB[2] = { (unsigned)(size_t)&Bs[0][srow][scol],
                       (unsigned)(size_t)&Bs[1][srow][scol] };
    unsigned long long gA = (unsigned long long)(A  + (size_t)(m0 + srow) * K + scol);
    unsigned long long gB = (unsigned long long)(Bt + (size_t)(n0 + srow) * K + scol);

    const int nk = K >> 5;  // K/32 stages
    async_copy_tile(lA[0], gA, lB[0], gB);

    for (int s = 0; s < nk; ++s) {
        const int buf = s & 1;
        if (s + 1 < nk) {
            gA += 64;  // 32 bf16 = 64 bytes per stage
            gB += 64;
            async_copy_tile(lA[buf ^ 1], gA, lB[buf ^ 1], gB);
            asm volatile("s_wait_asynccnt 0x4" ::: "memory");  // stage s landed
        } else {
            asm volatile("s_wait_asynccnt 0x0" ::: "memory");
        }
        __syncthreads();   // all waves' async writes for stage s visible

        v16bf af[2], bfv[4];
#pragma unroll
        for (int i = 0; i < 2; ++i)
            af[i] = frag_a(&As[buf][wm * 32 + i * 16 + lm][kbA]);
#pragma unroll
        for (int j = 0; j < 4; ++j)
            bfv[j] = frag_b(&Bs[buf][wn * 64 + j * 16 + lm][kbB]);

#pragma unroll
        for (int i = 0; i < 2; ++i)
#pragma unroll
            for (int j = 0; j < 4; ++j)
                acc[i][j] = wmma_bf16(af[i], bfv[j], acc[i][j]);

        __syncthreads();   // reads done -> buf may be overwritten next round
    }

#pragma unroll
    for (int i = 0; i < 2; ++i)
#pragma unroll
        for (int j = 0; j < 4; ++j)
#pragma unroll
            for (int r = 0; r < 8; ++r) {
                int row = m0 + wm * 32 + i * 16 + r + 8 * hh;
                int col = n0 + wn * 64 + j * 16 + lm;
                if (OUT_BF16) Cb[(size_t)row * N + col] = f2bf(acc[i][j][r]);
                else          Cf[(size_t)row * N + col] = acc[i][j][r];
            }
}

// ---------------------------------------------------------------------------
// Flash attention (causal), bf16 q/k/v laid out [T,B,H,DH] (= GEMM output
// [T*B, H*DH]). Block = 128 threads = 4 waves; wave owns 16 query rows;
// block covers 64 rows of one (b,h). QK^T and P*V via WMMA.
// ---------------------------------------------------------------------------
__global__ void __launch_bounds__(128)
flash_attn(const unsigned short* __restrict__ q,
           const unsigned short* __restrict__ k,
           const unsigned short* __restrict__ v,
           unsigned short* __restrict__ o) {
    constexpr int Bc = 8, Hh = 16, DHd = 64;
    constexpr float SCALE = 0.125f;  // 1/sqrt(64)

    __shared__ alignas(16) unsigned short vt[64][40];      // V^T tile [dim][key]
    __shared__ alignas(16) unsigned short pl[4][16][40];   // per-wave P tile

    const int tid  = threadIdx.x;
    const int lane = tid & 31;
    const int w    = tid >> 5;
    const int lm   = lane & 15;
    const int hh   = lane >> 4;
    const int kbA  = hh * 8;
    const int kbB  = hh * 8;
    const int bh   = blockIdx.y;
    const int b    = bh >> 4;           // H == 16
    const int h    = bh & 15;
    const int t0   = blockIdx.x * 64 + w * 16;

    // Q fragments (A-matrix, 16 rows x 64 dims -> 2 K-steps of 32)
    v16bf qf[2];
    {
        size_t base = (((size_t)(t0 + lm) * Bc + b) * Hh + h) * DHd;
        qf[0] = frag_a(q + base + kbA);
        qf[1] = frag_a(q + base + 32 + kbA);
    }

    float mrow[8], lrow[8];
    v8f   oacc[4];
#pragma unroll
    for (int r = 0; r < 8; ++r) { mrow[r] = -1e30f; lrow[r] = 0.0f; }
#pragma unroll
    for (int nt = 0; nt < 4; ++nt)
#pragma unroll
        for (int r = 0; r < 8; ++r) oacc[nt][r] = 0.0f;

    const int ntiles = blockIdx.x * 2 + 2;  // causal: up to block's last row
    for (int j = 0; j < ntiles; ++j) {
        const int kt0 = j * 32;
        __syncthreads();
        // stage V^T: vt[d][kk] = v[key kt0+kk, dim d]
        {
            int kk = tid >> 2;
            int d0 = (tid & 3) * 16;
            size_t vb = (((size_t)(kt0 + kk) * Bc + b) * Hh + h) * DHd + d0;
#pragma unroll
            for (int i = 0; i < 16; ++i)
                vt[d0 + i][kk] = v[vb + i];
        }
        __syncthreads();

        // S = Q K^T * SCALE for 32 keys (two 16-key chunks)
        v8f s[2];
#pragma unroll
        for (int c = 0; c < 2; ++c) {
#pragma unroll
            for (int r = 0; r < 8; ++r) s[c][r] = 0.0f;
            size_t kb2 = (((size_t)(kt0 + c * 16 + lm) * Bc + b) * Hh + h) * DHd;
#pragma unroll
            for (int ks = 0; ks < 2; ++ks) {
                v16bf kf = frag_b(k + kb2 + ks * 32 + kbB);
                s[c] = wmma_bf16(qf[ks], kf, s[c]);
            }
        }

        // online softmax per row (rows live across r; cols across 16 lanes)
#pragma unroll
        for (int r = 0; r < 8; ++r) {
            int   row = t0 + r + 8 * hh;
            float s0  = s[0][r] * SCALE;
            float s1  = s[1][r] * SCALE;
            if (kt0 + lm      > row) s0 = -1e30f;
            if (kt0 + 16 + lm > row) s1 = -1e30f;
            float mx = fmaxf(s0, s1);
#pragma unroll
            for (int d = 1; d < 16; d <<= 1) mx = fmaxf(mx, __shfl_xor(mx, d, 32));
            float mnew = fmaxf(mrow[r], mx);
            float corr = __expf(mrow[r] - mnew);
            mrow[r] = mnew;
            float p0 = __expf(s0 - mnew);
            float p1 = __expf(s1 - mnew);
            float ps = p0 + p1;
#pragma unroll
            for (int d = 1; d < 16; d <<= 1) ps += __shfl_xor(ps, d, 32);
            lrow[r] = lrow[r] * corr + ps;
#pragma unroll
            for (int nt = 0; nt < 4; ++nt) oacc[nt][r] *= corr;
            int prow = r + 8 * hh;
            pl[w][prow][lm]      = f2bf(p0);
            pl[w][prow][16 + lm] = f2bf(p1);
        }
        __syncthreads();

        // O += P (16x32) * V (32x64)
        v16bf pf = frag_a(&pl[w][lm][kbA]);
#pragma unroll
        for (int nt = 0; nt < 4; ++nt) {
            v16bf vf = frag_b(&vt[nt * 16 + lm][kbB]);
            oacc[nt] = wmma_bf16(pf, vf, oacc[nt]);
        }
    }

    // epilogue: normalize and store bf16 [T*B, H*DH]
#pragma unroll
    for (int nt = 0; nt < 4; ++nt)
#pragma unroll
        for (int r = 0; r < 8; ++r) {
            int t   = t0 + r + 8 * hh;
            int col = h * DHd + nt * 16 + lm;
            float val = oacc[nt][r] / lrow[r];
            o[(size_t)(t * Bc + b) * (Hh * DHd) + col] = f2bf(val);
        }
}

// ---------------------------------------------------------------------------
// LayerNorm over D=1024, output bf16. One block (256 thr) per row.
// ---------------------------------------------------------------------------
__global__ void __launch_bounds__(256)
layernorm_bf16(const float* __restrict__ x, const float* __restrict__ g,
               const float* __restrict__ bb, unsigned short* __restrict__ out) {
    constexpr int D = 1024;
    const int row = blockIdx.x;
    const float* xr = x + (size_t)row * D;
    float v4[4], s = 0.f, sq = 0.f;
#pragma unroll
    for (int i = 0; i < 4; ++i) {
        float t = xr[threadIdx.x + 256 * i];
        v4[i] = t; s += t; sq += t * t;
    }
#pragma unroll
    for (int d = 16; d >= 1; d >>= 1) {
        s  += __shfl_xor(s,  d, 32);
        sq += __shfl_xor(sq, d, 32);
    }
    __shared__ float rs_[8], rq_[8];
    if ((threadIdx.x & 31) == 0) { rs_[threadIdx.x >> 5] = s; rq_[threadIdx.x >> 5] = sq; }
    __syncthreads();
    if (threadIdx.x == 0) {
        float ts = 0.f, tq = 0.f;
        for (int i = 0; i < 8; ++i) { ts += rs_[i]; tq += rq_[i]; }
        float mu  = ts / D;
        float var = tq / D - mu * mu;
        rs_[0] = mu; rq_[0] = rsqrtf(var + 1e-5f);
    }
    __syncthreads();
    float mu = rs_[0], rstd = rq_[0];
#pragma unroll
    for (int i = 0; i < 4; ++i) {
        int c = threadIdx.x + 256 * i;
        out[(size_t)row * D + c] = f2bf((v4[i] - mu) * rstd * g[c] + bb[c]);
    }
}

// ---------------------------------------------------------------------------
// Elementwise / conversion kernels
// ---------------------------------------------------------------------------
__global__ void conv_bf16(const float* __restrict__ in,
                          unsigned short* __restrict__ out, long n) {
    long i = (long)blockIdx.x * 256 + threadIdx.x;
    if (i < n) out[i] = f2bf(in[i]);
}

// W[K,N] fp32 -> Wt[N,K] bf16
__global__ void conv_transpose_bf16(const float* __restrict__ W,
                                    unsigned short* __restrict__ Wt,
                                    int K, int N) {
    long idx = (long)blockIdx.x * 256 + threadIdx.x;
    if (idx >= (long)K * N) return;
    int kk = (int)(idx / N);
    int n  = (int)(idx % N);
    Wt[(size_t)n * K + kk] = f2bf(W[idx]);
}

__global__ void bias_relu_bf16(const float* __restrict__ in, const float* __restrict__ bias,
                               unsigned short* __restrict__ out, int N, long total) {
    long i = (long)blockIdx.x * 256 + threadIdx.x;
    if (i >= total) return;
    float t = in[i] + bias[i % N];
    out[i] = f2bf(t > 0.f ? t : 0.f);
}

__global__ void bias_bf16(const float* __restrict__ in, const float* __restrict__ bias,
                          unsigned short* __restrict__ out, int N, long total) {
    long i = (long)blockIdx.x * 256 + threadIdx.x;
    if (i >= total) return;
    out[i] = f2bf(in[i] + bias[i % N]);
}

// GRU gate combine:
//  r = sigmoid(Cr); z = sigmoid(Cz + bz - 0.1); h = tanh(Chw + r*Chu)
//  out = (1-z)*x + z*h
__global__ void gru_combine(const float* __restrict__ x,  const float* __restrict__ cr,
                            const float* __restrict__ cz, const float* __restrict__ chw,
                            const float* __restrict__ chu, const float* __restrict__ bz,
                            float* __restrict__ out, int Dd, long total) {
    long i = (long)blockIdx.x * 256 + threadIdx.x;
    if (i >= total) return;
    int d = (int)(i % Dd);
    float r = 1.f / (1.f + __expf(-cr[i]));
    float z = 1.f / (1.f + __expf(-(cz[i] + bz[d] - 0.1f)));
    float h = tanhf(chw[i] + r * chu[i]);
    float xv = x[i];
    out[i] = (1.f - z) * xv + z * h;
}

// ---------------------------------------------------------------------------
// Host orchestration
// ---------------------------------------------------------------------------
extern "C" void kernel_launch(void* const* d_in, const int* in_sizes, int n_in,
                              void* d_out, int out_size, void* d_ws, size_t ws_size,
                              hipStream_t stream) {
    (void)in_sizes; (void)n_in; (void)out_size; (void)ws_size;
    constexpr int T = 1024, Bc = 8, D = 1024, Hh = 16, DHd = 64, FF = 4096;
    constexpr int M = T * Bc;                 // 8192
    constexpr size_t DD = (size_t)D * D;

    const float* x     = (const float*)d_in[0];
    const float* Wq    = (const float*)d_in[1];
    const float* Wk    = (const float*)d_in[2];
    const float* Wv    = (const float*)d_in[3];
    const float* Wo    = (const float*)d_in[4];
    const float* ln1g  = (const float*)d_in[5];
    const float* ln1b  = (const float*)d_in[6];
    const float* W1    = (const float*)d_in[7];
    const float* b1    = (const float*)d_in[8];
    const float* W2    = (const float*)d_in[9];
    const float* b2    = (const float*)d_in[10];
    const float* ln2g  = (const float*)d_in[11];
    const float* ln2b  = (const float*)d_in[12];
    const float* g1W   = (const float*)d_in[13];
    const float* g1U   = (const float*)d_in[14];
    const float* g1bz  = (const float*)d_in[15];
    const float* g2W   = (const float*)d_in[16];
    const float* g2U   = (const float*)d_in[17];
    const float* g2bz  = (const float*)d_in[18];
    // d_in[19] = attn_mask (causal) — implemented analytically in flash_attn

    char* p = (char*)d_ws;
    auto alloc = [&](size_t bytes) {
        void* r = (void*)p;
        p += (bytes + 255) & ~(size_t)255;
        return r;
    };
    auto ubuf = [&](size_t elems) { return (unsigned short*)alloc(elems * 2); };
    auto fbuf = [&](size_t elems) { return (float*)alloc(elems * 4); };

    unsigned short* xbf   = ubuf((size_t)M * D);
    unsigned short* xnbf  = ubuf((size_t)M * D);
    unsigned short* qbf   = ubuf((size_t)M * D);
    unsigned short* kbf   = ubuf((size_t)M * D);
    unsigned short* vbf   = ubuf((size_t)M * D);
    unsigned short* abf   = ubuf((size_t)M * D);   // attention output
    float*          src2  = fbuf((size_t)M * D);
    unsigned short* ybf   = ubuf((size_t)M * D);   // bf16(src2)
    float*          Cr    = fbuf((size_t)M * D);
    float*          Cz    = fbuf((size_t)M * D);
    float*          Chw   = fbuf((size_t)M * D);
    float*          Chu   = fbuf((size_t)M * D);
    float*          srcf  = fbuf((size_t)M * D);
    unsigned short* snbf  = ubuf((size_t)M * D);
    float*          ff1   = fbuf((size_t)M * FF);
    unsigned short* ff1bf = ubuf((size_t)M * FF);
    float*          ff2   = fbuf((size_t)M * D);
    unsigned short* y2bf  = ubuf((size_t)M * D);
    unsigned short* sbf   = ubuf((size_t)M * D);   // bf16(srcf)
    unsigned short* WqT   = ubuf(DD);
    unsigned short* WkT   = ubuf(DD);
    unsigned short* WvT   = ubuf(DD);
    unsigned short* WoT   = ubuf(DD);
    unsigned short* W1T   = ubuf((size_t)D * FF);
    unsigned short* W2T   = ubuf((size_t)D * FF);
    unsigned short* g1WT  = ubuf(3 * DD);
    unsigned short* g1UT  = ubuf(3 * DD);
    unsigned short* g2WT  = ubuf(3 * DD);
    unsigned short* g2UT  = ubuf(3 * DD);

    auto conv_T = [&](const float* W, unsigned short* Wt, int K, int N) {
        long tot = (long)K * N;
        conv_transpose_bf16<<<(unsigned)((tot + 255) / 256), 256, 0, stream>>>(W, Wt, K, N);
    };
    auto gemm_bf = [&](const unsigned short* A, const unsigned short* Bt,
                       unsigned short* Cb, int Mm, int Nn, int Kk) {
        dim3 g(Mm / 128, Nn / 128);
        wmma_gemm<true, false><<<g, 256, 0, stream>>>(A, Bt, nullptr, Cb, Mm, Nn, Kk);
    };
    auto gemm_f = [&](const unsigned short* A, const unsigned short* Bt,
                      float* Cf, int Mm, int Nn, int Kk, bool accum) {
        dim3 g(Mm / 128, Nn / 128);
        if (accum) wmma_gemm<false, true ><<<g, 256, 0, stream>>>(A, Bt, Cf, nullptr, Mm, Nn, Kk);
        else       wmma_gemm<false, false><<<g, 256, 0, stream>>>(A, Bt, Cf, nullptr, Mm, Nn, Kk);
    };

    // ---- weight precision conversion (fp32 -> bf16, transposed to [N][K]) ----
    conv_T(Wq, WqT, D, D);
    conv_T(Wk, WkT, D, D);
    conv_T(Wv, WvT, D, D);
    conv_T(Wo, WoT, D, D);
    conv_T(W1, W1T, D, FF);
    conv_T(W2, W2T, FF, D);
    for (int i = 0; i < 3; ++i) {
        conv_T(g1W + i * DD, g1WT + i * DD, D, D);
        conv_T(g1U + i * DD, g1UT + i * DD, D, D);
        conv_T(g2W + i * DD, g2WT + i * DD, D, D);
        conv_T(g2U + i * DD, g2UT + i * DD, D, D);
    }

    const long MD = (long)M * D;
    const unsigned ew = (unsigned)((MD + 255) / 256);

    // ---- attention block (pre-LN) ----
    conv_bf16<<<ew, 256, 0, stream>>>(x, xbf, MD);
    layernorm_bf16<<<M, 256, 0, stream>>>(x, ln1g, ln1b, xnbf);
    gemm_bf(xnbf, WqT, qbf, M, D, D);
    gemm_bf(xnbf, WkT, kbf, M, D, D);
    gemm_bf(xnbf, WvT, vbf, M, D, D);
    flash_attn<<<dim3(T / 64, Bc * Hh), 128, 0, stream>>>(qbf, kbf, vbf, abf);
    gemm_f(abf, WoT, src2, M, D, D, false);

    // ---- GRU gate 1: src = gru(x, src2) ----
    conv_bf16<<<ew, 256, 0, stream>>>(src2, ybf, MD);
    gemm_f(ybf, g1WT + 0 * DD, Cr,  M, D, D, false);
    gemm_f(xbf, g1UT + 0 * DD, Cr,  M, D, D, true);
    gemm_f(ybf, g1WT + 1 * DD, Cz,  M, D, D, false);
    gemm_f(xbf, g1UT + 1 * DD, Cz,  M, D, D, true);
    gemm_f(ybf, g1WT + 2 * DD, Chw, M, D, D, false);
    gemm_f(xbf, g1UT + 2 * DD, Chu, M, D, D, false);
    gru_combine<<<ew, 256, 0, stream>>>(x, Cr, Cz, Chw, Chu, g1bz, srcf, D, MD);

    // ---- FFN (pre-LN) ----
    layernorm_bf16<<<M, 256, 0, stream>>>(srcf, ln2g, ln2b, snbf);
    gemm_f(snbf, W1T, ff1, M, FF, D, false);
    const long MF = (long)M * FF;
    bias_relu_bf16<<<(unsigned)((MF + 255) / 256), 256, 0, stream>>>(ff1, b1, ff1bf, FF, MF);
    gemm_f(ff1bf, W2T, ff2, M, D, FF, false);
    bias_bf16<<<ew, 256, 0, stream>>>(ff2, b2, y2bf, D, MD);

    // ---- GRU gate 2: out = gru(src, ff) ----
    conv_bf16<<<ew, 256, 0, stream>>>(srcf, sbf, MD);
    gemm_f(y2bf, g2WT + 0 * DD, Cr,  M, D, D, false);
    gemm_f(sbf,  g2UT + 0 * DD, Cr,  M, D, D, true);
    gemm_f(y2bf, g2WT + 1 * DD, Cz,  M, D, D, false);
    gemm_f(sbf,  g2UT + 1 * DD, Cz,  M, D, D, true);
    gemm_f(y2bf, g2WT + 2 * DD, Chw, M, D, D, false);
    gemm_f(sbf,  g2UT + 2 * DD, Chu, M, D, D, false);
    gru_combine<<<ew, 256, 0, stream>>>(srcf, Cr, Cz, Chw, Chu, g2bz,
                                        (float*)d_out, D, MD);
}
